// FFPTrader_61503931679125
// MI455X (gfx1250) — matrix operand order repaired and checked by reading, not compiled, not run on hardware
//
#include <hip/hip_runtime.h>
#include <hip/hip_bf16.h>
#include <math.h>

typedef __attribute__((ext_vector_type(16))) __bf16 v16bf;
typedef __attribute__((ext_vector_type(8)))  float  v8f;

constexpr int NB = 256;   // batch
constexpr int NS = 512;   // sequence
constexpr int NH = 2;     // heads
// E = 4, DH = 2

struct Params {
  const float* in[52];
  float* embs;   // [NB*NS*4]
  float* qb;     // [NS*NH*NB*2]
  float* kb;     // [NS*NH*NB*2]
  float* vb;     // [NS*NH*NB*2]
  float* att;    // [NB*NS*4]
  float* pooled; // [NB*4]
  float* out;    // [NB*3 + NB]
};

__device__ __forceinline__ float gelu_exact(float x) {
  return 0.5f * x * (1.0f + erff(x * 0.70710678118654752f));
}

// ---------------------------------------------------------------------------
// Kernel 1: one wave32 per (b,s). Lane = conv output position.
// conv1(2->4,k4) -> gelu -> conv2(4->8,k4) -> gelu -> conv3(8->16,k2) -> gelu
// -> emb linear (368 conv feats + bidask + price + possession) -> gelu -> +PE
// -> QKV projection, scattered to [S][H][B][2] layout for attention.
// ---------------------------------------------------------------------------
__global__ __launch_bounds__(256) void k_featurize(Params P) {
  __shared__ float a1s[8][4][27];
  __shared__ float a2s[8][8][24];
  const int w    = threadIdx.x >> 5;
  const int lane = threadIdx.x & 31;
  const int wid  = blockIdx.x * 8 + w;
  const int b    = wid >> 9;          // / NS
  const int s    = wid & (NS - 1);

  const float* ob  = P.in[0] + (size_t)(b * NS + s) * 2 * 30;
  // pull the wave's 240B input window toward the WGP (global_prefetch_b8)
  if (lane < 2) __builtin_prefetch(ob + lane * 32, 0, 0);

  const float* c1w = P.in[4]; const float* c1b = P.in[5];
  const float* c2w = P.in[6]; const float* c2b = P.in[7];
  const float* c3w = P.in[8]; const float* c3b = P.in[9];

  if (lane < 27) {
    float x0[4], x1[4];
#pragma unroll
    for (int k = 0; k < 4; k++) { x0[k] = ob[lane + k]; x1[k] = ob[30 + lane + k]; }
#pragma unroll
    for (int oc = 0; oc < 4; oc++) {
      float a = c1b[oc];
#pragma unroll
      for (int k = 0; k < 4; k++) a += x0[k] * c1w[oc*8 + k] + x1[k] * c1w[oc*8 + 4 + k];
      a1s[w][oc][lane] = gelu_exact(a);
    }
  }
  __syncthreads();

  if (lane < 24) {
#pragma unroll
    for (int oc = 0; oc < 8; oc++) {
      float a = c2b[oc];
#pragma unroll
      for (int ic = 0; ic < 4; ic++)
#pragma unroll
        for (int k = 0; k < 4; k++) a += a1s[w][ic][lane + k] * c2w[oc*16 + ic*4 + k];
      a2s[w][oc][lane] = gelu_exact(a);
    }
  }
  __syncthreads();

  float acc[4] = {0.f, 0.f, 0.f, 0.f};
  const float* ew = P.in[18];
  if (lane < 23) {
#pragma unroll
    for (int oc = 0; oc < 16; oc++) {
      float v = c3b[oc];
#pragma unroll
      for (int ic = 0; ic < 8; ic++)
        v += a2s[w][ic][lane] * c3w[oc*16 + ic*2] + a2s[w][ic][lane + 1] * c3w[oc*16 + ic*2 + 1];
      v = gelu_exact(v);
      const int f = oc * 23 + lane;   // orders feature index
#pragma unroll
      for (int e = 0; e < 4; e++) acc[e] += v * ew[e*371 + f];
    }
  }
  // wave32 reduction of the emb partial dot products
#pragma unroll
  for (int off = 16; off > 0; off >>= 1)
#pragma unroll
    for (int e = 0; e < 4; e++) acc[e] += __shfl_xor(acc[e], off, 32);

  if (lane == 0) {
    const float* eb = P.in[19];
    const float t0 = P.in[1][(size_t)(b*NS + s)*2 + 0];
    const float t1 = P.in[1][(size_t)(b*NS + s)*2 + 1];
    const float* baw1 = P.in[10]; const float* bab1 = P.in[11];
    float h0 = gelu_exact(baw1[0]*t0 + baw1[1]*t1 + bab1[0]);
    float h1 = gelu_exact(baw1[2]*t0 + baw1[3]*t1 + bab1[1]);
    float bidask = gelu_exact(P.in[12][0]*h0 + P.in[12][1]*h1 + P.in[13][0]);
    float pr = P.in[2][b*NS + s];
    float p1 = gelu_exact(P.in[14][0]*pr + P.in[15][0]);
    float price = gelu_exact(P.in[16][0]*p1 + P.in[17][0]);
    float poss = P.in[3][b*NS + s];

    const float sf = (float)s;
    float pe[4];
    pe[0] = sinf(sf) * 0.01f;        pe[1] = cosf(sf) * 0.01f;
    pe[2] = sinf(sf*0.01f) * 0.01f;  pe[3] = cosf(sf*0.01f) * 0.01f;

    float emb[4];
#pragma unroll
    for (int e = 0; e < 4; e++) {
      float a = acc[e] + ew[e*371+368]*bidask + ew[e*371+369]*price + ew[e*371+370]*poss + eb[e];
      emb[e] = gelu_exact(a) + pe[e];
      P.embs[(size_t)(b*NS + s)*4 + e] = emb[e];
    }
    const float* inw = P.in[20]; const float* inb = P.in[21];
    float qkv[12];
#pragma unroll
    for (int j = 0; j < 12; j++) {
      float a = inb[j];
#pragma unroll
      for (int e = 0; e < 4; e++) a += inw[j*4 + e] * emb[e];
      qkv[j] = a;
    }
#pragma unroll
    for (int hh = 0; hh < 2; hh++)
#pragma unroll
      for (int d = 0; d < 2; d++) {
        size_t base = (((size_t)s*NH + hh)*NB + b)*2 + d;
        P.qb[base] = qkv[hh*2 + d];
        P.kb[base] = qkv[4 + hh*2 + d];
        P.vb[base] = qkv[8 + hh*2 + d];
      }
  }
}

// ---------------------------------------------------------------------------
// Kernel 2: attention over the batch axis. One block per (s,h).
// Two-pass softmax stats (K=2 dot is cheap to recompute), then O = A @ V as
// bf16 WMMA 16x16x32: M=256 queries, K=256 keys (8 chunks of 32), N=2 pad 16.
// A staged unnormalized (exp) in bf16 LDS; 1/rowsum folded into the output.
// ---------------------------------------------------------------------------
__global__ __launch_bounds__(256) void k_attn(Params P) {
  __shared__ float qs[NB][2], ks[NB][2], vs[NB][2];
  __shared__ float rsinv[NB];
  __shared__ __bf16 Ac[NB][32];     // exp-score staging chunk (rows x 32 keys)

  const int s = blockIdx.x >> 1;
  const int h = blockIdx.x & 1;
  const int t = threadIdx.x;

  const size_t base = (((size_t)s*NH + h)*NB)*2;
  qs[t][0] = P.qb[base + t*2]; qs[t][1] = P.qb[base + t*2 + 1];
  ks[t][0] = P.kb[base + t*2]; ks[t][1] = P.kb[base + t*2 + 1];
  vs[t][0] = P.vb[base + t*2]; vs[t][1] = P.vb[base + t*2 + 1];
  __syncthreads();

  const float scale = 0.70710678118654752f;   // 1/sqrt(DH=2)
  const float qx = qs[t][0], qy = qs[t][1];
  float mx = -1e30f;
  for (int m = 0; m < NB; m++) {
    float sc = (qx*ks[m][0] + qy*ks[m][1]) * scale;
    mx = fmaxf(mx, sc);
  }
  float sum = 0.f;
  for (int m = 0; m < NB; m++) {
    float sc = (qx*ks[m][0] + qy*ks[m][1]) * scale;
    sum += __expf(sc - mx);
  }
  rsinv[t] = 1.0f / sum;
  __syncthreads();

  const int lane  = t & 31;
  const int wv    = t >> 5;
  const int n     = lane & 15;             // output column (DH pad to 16)
  const int khalf = (lane >> 4) * 8;       // A-fragment K split per ISA layout
  const int kof   = (lane >> 4) * 16;      // unused; B fragment uses the khalf split, as A
  v8f c0 = {}; v8f c1 = {};

  for (int kk = 0; kk < 8; kk++) {
    // stage exp-scores for this row's 32-key chunk (bf16, unnormalized)
#pragma unroll
    for (int j = 0; j < 32; j++) {
      int m = kk*32 + j;
      float sc = (qx*ks[m][0] + qy*ks[m][1]) * scale;
      Ac[t][j] = (__bf16)__expf(sc - mx);
    }
    __syncthreads();

    // B fragment (32x16): lane holds column n; K order matches A (khalf+j, khalf+16+j)
    v16bf bf;
#pragma unroll
    for (int j = 0; j < 16; j++) {
      int kg = kk*32 + khalf + (j & 7) + ((j >> 3) << 4);
      bf[j] = (n < 2) ? (__bf16)vs[kg][n] : (__bf16)0.0f;
    }
    // two 16-row M tiles per wave
    {
      int row = (2*wv)*16 + (lane & 15);
      v16bf af;
#pragma unroll
      for (int j = 0; j < 8; j++) { af[j] = Ac[row][khalf + j]; af[8 + j] = Ac[row][khalf + 16 + j]; }
      c0 = __builtin_amdgcn_wmma_f32_16x16x32_bf16(false, af, false, bf, (short)0, c0, false, false);
    }
    {
      int row = (2*wv + 1)*16 + (lane & 15);
      v16bf af;
#pragma unroll
      for (int j = 0; j < 8; j++) { af[j] = Ac[row][khalf + j]; af[8 + j] = Ac[row][khalf + 16 + j]; }
      c1 = __builtin_amdgcn_wmma_f32_16x16x32_bf16(false, af, false, bf, (short)0, c1, false, false);
    }
    __syncthreads();
  }

  // D layout: VGPR g -> row = tile*16 + g + 8*(lane/16), col n = lane%16
  if (n < 2) {
#pragma unroll
    for (int g = 0; g < 8; g++) {
      int r0 = (2*wv)*16 + g + ((lane >> 4) << 3);
      P.att[((size_t)r0*NS + s)*4 + h*2 + n] = c0[g] * rsinv[r0];
      int r1 = (2*wv + 1)*16 + g + ((lane >> 4) << 3);
      P.att[((size_t)r1*NS + s)*4 + h*2 + n] = c1[g] * rsinv[r1];
    }
  }
}

// ---------------------------------------------------------------------------
// Kernel 3: per-token out-proj + residual + LN1 + FFN(relu) + residual + LN2.
// Mean-pool via hierarchical reduction: wave32 shfl -> LDS -> 4 atomics/block
// (each block's 256 tokens all belong to one batch row b = blockIdx/2).
// ---------------------------------------------------------------------------
__global__ __launch_bounds__(256) void k_post(Params P) {
  __shared__ float wpart[8][4];
  const int idx = blockIdx.x * 256 + threadIdx.x;   // (b,s) flat
  const int b = blockIdx.x >> 1;                    // whole block shares b
  const float* ow  = P.in[22]; const float* obi = P.in[23];

  const float4 emv = ((const float4*)P.embs)[idx];
  const float4 atv = ((const float4*)P.att)[idx];
  float em[4] = {emv.x, emv.y, emv.z, emv.w};
  float at[4] = {atv.x, atv.y, atv.z, atv.w};
  float x[4];
#pragma unroll
  for (int e = 0; e < 4; e++) {
    float o = obi[e];
#pragma unroll
    for (int j = 0; j < 4; j++) o += ow[e*4 + j] * at[j];
    x[e] = em[e] + o;
  }
  // LN1
  float mu = 0.f;
#pragma unroll
  for (int e = 0; e < 4; e++) mu += x[e];
  mu *= 0.25f;
  float var = 0.f;
#pragma unroll
  for (int e = 0; e < 4; e++) { float d = x[e] - mu; var += d*d; }
  var *= 0.25f;
  float inv = rsqrtf(var + 1e-5f);
  float x1[4];
#pragma unroll
  for (int e = 0; e < 4; e++) x1[e] = (x[e] - mu) * inv * P.in[24][e] + P.in[25][e];
  // FFN 4 -> 8 (relu) -> 4
  float hd[8];
#pragma unroll
  for (int i = 0; i < 8; i++) {
    float a = P.in[29][i];
#pragma unroll
    for (int e = 0; e < 4; e++) a += P.in[28][i*4 + e] * x1[e];
    hd[i] = fmaxf(a, 0.f);
  }
  float x2[4];
#pragma unroll
  for (int e = 0; e < 4; e++) {
    float a = P.in[31][e];
#pragma unroll
    for (int i = 0; i < 8; i++) a += P.in[30][e*8 + i] * hd[i];
    x2[e] = x1[e] + a;
  }
  // LN2
  mu = 0.f;
#pragma unroll
  for (int e = 0; e < 4; e++) mu += x2[e];
  mu *= 0.25f;
  var = 0.f;
#pragma unroll
  for (int e = 0; e < 4; e++) { float d = x2[e] - mu; var += d*d; }
  var *= 0.25f;
  inv = rsqrtf(var + 1e-5f);
  float y[4];
#pragma unroll
  for (int e = 0; e < 4; e++) y[e] = (x2[e] - mu) * inv * P.in[26][e] + P.in[27][e];

  // hierarchical mean-pool reduction
#pragma unroll
  for (int off = 16; off > 0; off >>= 1)
#pragma unroll
    for (int e = 0; e < 4; e++) y[e] += __shfl_xor(y[e], off, 32);
  const int lane = threadIdx.x & 31;
  const int wv   = threadIdx.x >> 5;
  if (lane == 0) {
#pragma unroll
    for (int e = 0; e < 4; e++) wpart[wv][e] = y[e];
  }
  __syncthreads();
  if (threadIdx.x < 4) {
    float a = 0.f;
#pragma unroll
    for (int ww = 0; ww < 8; ww++) a += wpart[ww][threadIdx.x];
    atomicAdd(&P.pooled[b*4 + threadIdx.x], a * (1.0f / 512.0f));
  }
}

// ---------------------------------------------------------------------------
// Kernel 4: policy / value heads. One thread per batch row.
// ---------------------------------------------------------------------------
__global__ __launch_bounds__(256) void k_heads(Params P) {
  const int b = threadIdx.x;
  float base[4];
#pragma unroll
  for (int e = 0; e < 4; e++) base[e] = P.pooled[b*4 + e];

  const int mw[4] = {32, 34, 36, 38}, mb[4] = {33, 35, 37, 39};
  float cur[4];
#pragma unroll
  for (int e = 0; e < 4; e++) cur[e] = base[e];
  for (int l = 0; l < 4; l++) {
    const float* W = P.in[mw[l]]; const float* Bv = P.in[mb[l]];
    float nh[4];
#pragma unroll
    for (int i = 0; i < 4; i++) {
      float a = Bv[i];
#pragma unroll
      for (int j = 0; j < 4; j++) a += W[i*4 + j] * cur[j];
      nh[i] = gelu_exact(a);
    }
#pragma unroll
    for (int i = 0; i < 4; i++) cur[i] = nh[i];
  }
#pragma unroll
  for (int r = 0; r < 3; r++) {
    float a = P.in[41][r];
#pragma unroll
    for (int j = 0; j < 4; j++) a += P.in[40][r*4 + j] * cur[j];
    P.out[b*3 + r] = a;
  }

  const int vw[4] = {42, 44, 46, 48}, vb[4] = {43, 45, 47, 49};
#pragma unroll
  for (int e = 0; e < 4; e++) cur[e] = base[e];
  for (int l = 0; l < 4; l++) {
    const float* W = P.in[vw[l]]; const float* Bv = P.in[vb[l]];
    float nh[4];
#pragma unroll
    for (int i = 0; i < 4; i++) {
      float a = Bv[i];
#pragma unroll
      for (int j = 0; j < 4; j++) a += W[i*4 + j] * cur[j];
      nh[i] = gelu_exact(a);
    }
#pragma unroll
    for (int i = 0; i < 4; i++) cur[i] = nh[i];
  }
  float a = P.in[51][0];
#pragma unroll
  for (int j = 0; j < 4; j++) a += P.in[50][j] * cur[j];
  P.out[NB*3 + b] = a;
}

extern "C" void kernel_launch(void* const* d_in, const int* in_sizes, int n_in,
                              void* d_out, int out_size, void* d_ws, size_t ws_size,
                              hipStream_t stream) {
  (void)in_sizes; (void)n_in; (void)out_size; (void)ws_size;
  Params P;
  for (int i = 0; i < 52; i++) P.in[i] = (const float*)d_in[i];
  float* ws = (float*)d_ws;
  size_t off = 0;
  P.embs   = ws + off; off += (size_t)NB*NS*4;      // 524288
  P.qb     = ws + off; off += (size_t)NS*NH*NB*2;   // 524288
  P.kb     = ws + off; off += (size_t)NS*NH*NB*2;
  P.vb     = ws + off; off += (size_t)NS*NH*NB*2;
  P.att    = ws + off; off += (size_t)NB*NS*4;
  P.pooled = ws + off; off += (size_t)NB*4;
  P.out    = (float*)d_out;

  hipMemsetAsync(P.pooled, 0, (size_t)NB*4*sizeof(float), stream);
  k_featurize<<<(NB*NS)/8, 256, 0, stream>>>(P);
  k_attn<<<NS*NH, 256, 0, stream>>>(P);
  k_post<<<(NB*NS)/256, 256, 0, stream>>>(P);
  k_heads<<<1, 256, 0, stream>>>(P);
}
